// LogicGatedSNN_8787503088069
// MI455X (gfx1250) — matrix operand order repaired and checked by reading, not compiled, not run on hardware
//
#include <hip/hip_runtime.h>

typedef __attribute__((ext_vector_type(16))) _Float16 v16h;
typedef __attribute__((ext_vector_type(8)))  float    v8f;
typedef __attribute__((ext_vector_type(4)))  float    v4f;

#define IN_F   8192
#define OUT_F  8192
#define KSPLIT 8
#define KLEN   (IN_F / KSPLIT)   // 1024
#define STATE_THRESH 50.0f

// ---------------- kernel 0: x = clip(spike_input * 2, 0, 1) ----------------
__global__ void snn_prep_x(const float* __restrict__ spike_input,
                           float* __restrict__ x) {
  int i = blockIdx.x * blockDim.x + threadIdx.x;
  if (i < IN_F) {
    float v = spike_input[i] * 2.0f;
    x[i] = fminf(fmaxf(v, 0.0f), 1.0f);
  }
}

// ---------------- kernel 1: current partials via WMMA ----------------------
// Each wave owns 16 output rows and one K-slice of length KLEN.
// A = (states > 50) as f16 0/1, 16x32 tile per step.
// B = x chunk broadcast into all 16 columns -> every column of D holds the
//     same 16 row partial sums; lanes 0 and 16 (N==0) write them out.
__global__ void __launch_bounds__(256) snn_matvec_wmma(
    const float* __restrict__ states,
    const float* __restrict__ x,
    float* __restrict__ partial) {
  const int lane = threadIdx.x & 31;
  const int wave = threadIdx.x >> 5;
  const int m    = lane & 15;   // row within 16-row tile (A: M = lane%16)
  const int half = lane >> 4;   // lane half selects K sub-blocks
  const int r0   = blockIdx.x * 128 + wave * 16;
  const int kb   = blockIdx.y * KLEN;

  const float* __restrict__ row = states + (size_t)(r0 + m) * IN_F;

  v8f c = {};
  for (int k = kb; k < kb + KLEN; k += 32) {
    // A fragment: K = {k+8h .. k+8h+7} in a[0..7], {k+16+8h ..} in a[8..15]
    const int ka = k + 8 * half;
    const int kc = k + 16 + 8 * half;
    v4f f0 = __builtin_nontemporal_load((const v4f*)(row + ka));
    v4f f1 = __builtin_nontemporal_load((const v4f*)(row + ka + 4));
    v4f f2 = __builtin_nontemporal_load((const v4f*)(row + kc));
    v4f f3 = __builtin_nontemporal_load((const v4f*)(row + kc + 4));
    v16h a;
#pragma unroll
    for (int j = 0; j < 4; ++j) {
      a[j]      = (_Float16)(f0[j] > STATE_THRESH ? 1.0f : 0.0f);
      a[4 + j]  = (_Float16)(f1[j] > STATE_THRESH ? 1.0f : 0.0f);
      a[8 + j]  = (_Float16)(f2[j] > STATE_THRESH ? 1.0f : 0.0f);
      a[12 + j] = (_Float16)(f3[j] > STATE_THRESH ? 1.0f : 0.0f);
    }
    // B fragment: K = {k+8h .. k+8h+7} in b[0..7], {k+16+8h ..} in b[8..15], same for every N
    const float* __restrict__ xb = x + k + 8 * half;
    v16h b;
#pragma unroll
    for (int j = 0; j < 16; ++j) b[j] = (_Float16)xb[j < 8 ? j : j + 8];

    c = __builtin_amdgcn_wmma_f32_16x16x32_f16(
        /*neg_a=*/false, a, /*neg_b=*/false, b,
        /*c_mod=*/(short)0, c, /*reuse_a=*/false, /*reuse_b=*/false);
  }

  // D layout: VGPR v, lane l -> D[M = v + 8*(l/16)][N = l%16]; all N equal.
  if (m == 0) {
    float* __restrict__ p = partial + (size_t)blockIdx.y * OUT_F + r0 + 8 * half;
#pragma unroll
    for (int v = 0; v < 8; ++v) p[v] = c[v];
  }
}

// ---------------- kernel 2: reduce partials + neuron dynamics --------------
__global__ void snn_neuron(const float* __restrict__ partial,
                           const float* __restrict__ mp,
                           const float* __restrict__ thr,
                           float* __restrict__ out) {
  int o = blockIdx.x * blockDim.x + threadIdx.x;
  if (o >= OUT_F) return;
  float cur = 0.0f;
#pragma unroll
  for (int s = 0; s < KSPLIT; ++s) cur += partial[(size_t)s * OUT_F + o];
  float v   = mp[o] * 0.7f + cur;
  float t   = thr[o];
  float spk = (v >= t) ? 1.0f : 0.0f;
  float nt  = fminf(fmaxf(t + (spk - 0.1f) * 0.05f, 0.5f), 5.0f);
  float nv  = v * (1.0f - spk) * 0.3f;
  out[o]             = spk;  // spikes
  out[OUT_F + o]     = nv;   // new_v_mem
  out[2 * OUT_F + o] = nt;   // new_threshold
}

// ---------------- kernel 3: streaming trace update (dominant cost) ---------
// new_trace = clip(trace*0.8 + spikes[o]*x[i], 0, 3); 512 MB of NT traffic.
__global__ void __launch_bounds__(256) snn_trace(
    const float* __restrict__ trace,
    const float* __restrict__ x,
    const float* __restrict__ spikes,
    float* __restrict__ ntrace) {
  const size_t n4     = (size_t)OUT_F * IN_F / 4;       // 16M float4s
  const size_t stride = (size_t)gridDim.x * blockDim.x;
  for (size_t i = (size_t)blockIdx.x * blockDim.x + threadIdx.x;
       i < n4; i += stride) {
    v4f tr = __builtin_nontemporal_load((const v4f*)trace + i);
    v4f xv = *((const v4f*)x + (i & (IN_F / 4 - 1)));
    float spk = spikes[i >> 11];                        // i / (IN_F/4)
    v4f r = tr * 0.8f + spk * xv;
#pragma unroll
    for (int j = 0; j < 4; ++j) r[j] = fminf(fmaxf(r[j], 0.0f), 3.0f);
    __builtin_nontemporal_store(r, (v4f*)ntrace + i);
  }
}

extern "C" void kernel_launch(void* const* d_in, const int* in_sizes, int n_in,
                              void* d_out, int out_size, void* d_ws, size_t ws_size,
                              hipStream_t stream) {
  (void)in_sizes; (void)n_in; (void)out_size; (void)ws_size;
  const float* spike_input = (const float*)d_in[0];  // [8192]
  const float* states      = (const float*)d_in[1];  // [8192,8192]
  const float* mp          = (const float*)d_in[2];  // [8192]
  const float* thr         = (const float*)d_in[3];  // [8192]
  const float* trace       = (const float*)d_in[4];  // [8192,8192]
  float* out = (float*)d_out;

  float* x       = (float*)d_ws;   // [IN_F]
  float* partial = x + IN_F;       // [KSPLIT * OUT_F]

  snn_prep_x<<<IN_F / 256, 256, 0, stream>>>(spike_input, x);
  snn_matvec_wmma<<<dim3(OUT_F / 128, KSPLIT), 256, 0, stream>>>(states, x, partial);
  snn_neuron<<<OUT_F / 256, 256, 0, stream>>>(partial, mp, thr, out);
  snn_trace<<<16384, 256, 0, stream>>>(trace, x, /*spikes=*/out,
                                       /*ntrace=*/out + 3 * OUT_F);
}
